// SCat_74964359184689
// MI455X (gfx1250) — compile-verified, matrix operand order checked by reading, not run on hardware
//
#include <hip/hip_runtime.h>

typedef __attribute__((ext_vector_type(16))) _Float16     v16h;
typedef __attribute__((ext_vector_type(8)))  float        v8f;
typedef __attribute__((ext_vector_type(4)))  unsigned int u32x4;

union Frag {
  v16h h;
  unsigned int u[8];
  u32x4 q[2];
};

__device__ __forceinline__ unsigned int pack2(float a, float b) {
  // v_cvt_pk_rtz_f16_f32
  return __builtin_bit_cast(unsigned int, __builtin_amdgcn_cvt_pkrtz(a, b));
}

// ---------------- problem sizes ----------------
constexpr int BSZ  = 4;
constexpr int CIN  = 512;
constexpr int CH   = 256;   // per-branch channels
constexpr int NPOS = 4096;  // H*W
constexpr int COUT = 512;

// ---------------- workspace layout (bytes) ----------------
constexpr size_t MB = 1ull << 20;
constexpr size_t OFF_Q1F  = 0;                    // [B][256 ntile][lane][8] u32 f16x2 : 1MB
constexpr size_t OFF_K1F  = 1 * MB;
constexpr size_t OFF_Q2F  = 2 * MB;
constexpr size_t OFF_K2F  = 3 * MB;
constexpr size_t OFF_V1F  = 4 * MB;               // [B][16 ct][128 mt][lane][8] : 8MB
constexpr size_t OFF_V2F  = 12 * MB;              // 8MB
constexpr size_t OFF_OUT  = 20 * MB;              // f32 [B][512][4096] : 32MB
constexpr size_t OFF_WQ1F = 52 * MB;              // 16KB each
constexpr size_t OFF_WK1F = 52 * MB + 16 * 1024;
constexpr size_t OFF_WQ2F = 52 * MB + 32 * 1024;
constexpr size_t OFF_WK2F = 52 * MB + 48 * 1024;
constexpr size_t OFF_WV1F = 52 * MB + 64 * 1024;  // 128KB each
constexpr size_t OFF_WV2F = 52 * MB + 192 * 1024;
constexpr size_t OFF_WCF  = 52 * MB + 320 * 1024; // 512KB

// =====================================================================
// Weight -> f16 A-fragment layout.  Frag dword index: fid*256 + lane*8 + d
// element (o, c): lane = (o&15) + 16*((c>>3)&1), d = ((c>>1)&3)+4*((c>>4)&1)
// =====================================================================
__global__ __launch_bounds__(256) void wfrag_kernel(const float* __restrict__ W,
                                                    unsigned int* __restrict__ dst,
                                                    int O, int Cw) {
  (void)O;
  const int t    = blockIdx.x * 256 + threadIdx.x;
  const int fid  = t >> 8;
  const int lane = (t >> 3) & 31;
  const int d    = t & 7;
  const int nct  = Cw >> 5;
  const int ct   = fid % nct;
  const int ot   = fid / nct;
  const int g    = lane >> 4;
  const int o    = ot * 16 + (lane & 15);
  const int c    = ct * 32 + 2 * (d & 3) + 8 * g + 16 * (d >> 2);
  dst[t] = pack2(W[(size_t)o * Cw + c], W[(size_t)o * Cw + c + 1]);
}

// =====================================================================
// Projection: q/k/v = W * x  (per branch).  One wave per (b, branch, n-tile16).
// K-loop over C=256 in 32-chunks; 20 WMMAs per chunk (2 q-tiles, 2 k-tiles,
// 16 v-tiles).  Writes q/k frags (lane-local repack) and v frags (scatter).
// =====================================================================
__global__ __launch_bounds__(256) void proj_kernel(
    const float* __restrict__ x,
    const float* __restrict__ bq1, const float* __restrict__ bk1, const float* __restrict__ bv1,
    const float* __restrict__ bq2, const float* __restrict__ bk2, const float* __restrict__ bv2,
    unsigned char* __restrict__ ws) {
  const int lane  = threadIdx.x & 31;
  const int wid   = blockIdx.x * 8 + (threadIdx.x >> 5);
  const int ntile = wid & 255;
  const int bh    = wid >> 8;
  const int b     = bh >> 1;
  const int hlf   = bh & 1;
  const int g     = lane >> 4;
  const int ln    = lane & 15;
  const int n     = ntile * 16 + ln;

  const float* xb = x + (size_t)(b * 2 + hlf) * CH * NPOS;
  const unsigned int* wq = (const unsigned int*)(ws + (hlf ? OFF_WQ2F : OFF_WQ1F));
  const unsigned int* wk = (const unsigned int*)(ws + (hlf ? OFF_WK2F : OFF_WK1F));
  const unsigned int* wv = (const unsigned int*)(ws + (hlf ? OFF_WV2F : OFF_WV1F));
  const float* bq = hlf ? bq2 : bq1;
  const float* bk = hlf ? bk2 : bk1;
  const float* bv = hlf ? bv2 : bv1;

  const v8f zeroV = {0.f, 0.f, 0.f, 0.f, 0.f, 0.f, 0.f, 0.f};
  v8f acc[20];
#pragma unroll
  for (int t = 0; t < 20; ++t) acc[t] = zeroV;

  for (int ci = 0; ci < 8; ++ci) {
    Frag bf;  // B operand: x tile (c=32 rows x n=16 cols), gathered + cvt to f16
#pragma unroll
    for (int d = 0; d < 8; ++d) {
      const int c = ci * 32 + 2 * (d & 3) + 8 * g + 16 * (d >> 2);
      bf.u[d] = pack2(xb[(size_t)c * NPOS + n], xb[(size_t)(c + 1) * NPOS + n]);
    }
#pragma unroll
    for (int t = 0; t < 20; ++t) {
      const unsigned int* wb =
          (t < 2) ? (wq + (size_t)(t * 8 + ci) * 256)
        : (t < 4) ? (wk + (size_t)((t - 2) * 8 + ci) * 256)
                  : (wv + (size_t)((t - 4) * 8 + ci) * 256);
      Frag af;
      const u32x4* p = (const u32x4*)wb + lane * 2;
      af.q[0] = p[0];
      af.q[1] = p[1];
      acc[t] = __builtin_amdgcn_wmma_f32_16x16x32_f16(false, af.h, false, bf.h,
                                                      (short)0, acc[t], false, false);
    }
  }

  // ---- q/k fragment stores (lane-local: D row r+8g+16t -> frag dword (r>>1)+4t)
  unsigned int* qf = (unsigned int*)(ws + (hlf ? OFF_Q2F : OFF_Q1F)) + (size_t)(b * 256 + ntile) * 256;
  unsigned int* kf = (unsigned int*)(ws + (hlf ? OFF_K2F : OFF_K1F)) + (size_t)(b * 256 + ntile) * 256;
  Frag qd, kd;
#pragma unroll
  for (int d = 0; d < 8; ++d) {
    const int t  = d >> 2;
    const int r0 = 2 * (d & 3);
    const int o0 = t * 16 + r0 + 8 * g;
    qd.u[d] = pack2(acc[t][r0] + bq[o0], acc[t][r0 + 1] + bq[o0 + 1]);
    kd.u[d] = pack2(acc[t + 2][r0] + bk[o0], acc[t + 2][r0 + 1] + bk[o0 + 1]);
  }
  ((u32x4*)qf)[lane * 2 + 0] = qd.q[0];
  ((u32x4*)qf)[lane * 2 + 1] = qd.q[1];
  ((u32x4*)kf)[lane * 2 + 0] = kd.q[0];
  ((u32x4*)kf)[lane * 2 + 1] = kd.q[1];

  // ---- v fragment scatter: value (c, m=n) -> A-frag of tile (c>>4, m>>5)
  _Float16* vh = (_Float16*)(ws + (hlf ? OFF_V2F : OFF_V1F));
  const int mloc = n & 31;
  const int mt   = n >> 5;
  const int g2   = (mloc >> 3) & 1;
  const int d2   = ((mloc >> 1) & 3) + 4 * (mloc >> 4);
  const int h2   = mloc & 1;
#pragma unroll
  for (int t = 4; t < 20; ++t) {
    const int ct = t - 4;
    const size_t fid = (size_t)(b * 16 + ct) * 128 + mt;
#pragma unroll
    for (int r = 0; r < 8; ++r) {
      const int c     = r + 8 * g + 16 * ct;
      const int lane2 = (r + 8 * g) + 16 * g2;
      vh[(fid * 256 + (size_t)lane2 * 8 + d2) * 2 + h2] = (_Float16)(acc[t][r] + bv[c]);
    }
  }
}

// =====================================================================
// Flash attention with workgroup-shared, double-buffered LDS staging of K/V
// via CDNA5 async global->LDS DMA (GLOBAL_LOAD_ASYNC_TO_LDS_B128, ASYNCcnt).
// Block = 8 waves sharing (b, attn); each wave owns one query n-tile16.
// Per 32-key step the block stages 2 K-frags (2KB) + 16 V-frags (16KB) into
// LDS once (vs 8x redundant L2 reads); step t+1's DMA is issued before the
// 18 WMMAs of step t so the copy overlaps matrix math, with no VGPRs spent
// on staging data.
// =====================================================================
__global__ __launch_bounds__(256) void attn_kernel(unsigned char* __restrict__ ws) {
  __shared__ __align__(16) unsigned int smem[2][4608];  // [stage][512 K dwords | 4096 V dwords]

  const int tid   = threadIdx.x;
  const int lane  = tid & 31;
  const int wid   = blockIdx.x * 8 + (tid >> 5);
  const int ntile = wid & 255;
  const int ba    = wid >> 8;  // block-uniform
  const int b     = ba >> 1;
  const int at    = ba & 1;    // 0: q2*k1,v1 -> out1 ; 1: q1*k2,v2 -> out2

  const unsigned int* qf  = (const unsigned int*)(ws + (at ? OFF_Q1F : OFF_Q2F)) + (size_t)(b * 256 + ntile) * 256;
  const unsigned int* kfB = (const unsigned int*)(ws + (at ? OFF_K2F : OFF_K1F)) + (size_t)(b * 256) * 256;
  const unsigned int* vfB = (const unsigned int*)(ws + (at ? OFF_V2F : OFF_V1F)) + (size_t)(b * 16) * 128 * 256;
  float* outp = (float*)(ws + OFF_OUT) + ((size_t)b * CIN + (size_t)at * CH) * NPOS;

  Frag qa;
  {
    const u32x4* p = (const u32x4*)qf + lane * 2;
    qa.q[0] = p[0];
    qa.q[1] = p[1];
  }

  // ---- cooperative async staging: 1152 16B-chunks per step, 256 threads ----
  auto issue_async = [&](int mt, int stg) {
#pragma unroll
    for (int j = 0; j < 5; ++j) {
      const int cid = j * 256 + tid;
      if (j < 4 || tid < 128) {               // wave-uniform predicate
        const u32x4* src;
        if (cid < 128) {                      // K: 2 frags, contiguous 2KB
          src = (const u32x4*)(kfB + (size_t)mt * 512) + cid;
        } else {                              // V: 16 frags of 1KB
          const int c = cid - 128;
          src = (const u32x4*)(vfB + ((size_t)(c >> 6) * 128 + mt) * 256) + (c & 63);
        }
        // low 32 bits of a flat pointer into LDS == workgroup LDS byte offset
        const unsigned ldsoff = (unsigned)(uintptr_t)(&smem[stg][cid * 4]);
        asm volatile("global_load_async_to_lds_b128 %0, %1, off"
                     :: "v"(ldsoff), "v"((unsigned long long)(uintptr_t)src)
                     : "memory");
      }
    }
  };

  const v8f zeroV = {0.f, 0.f, 0.f, 0.f, 0.f, 0.f, 0.f, 0.f};
  v8f oacc[16];
#pragma unroll
  for (int t = 0; t < 16; ++t) oacc[t] = zeroV;
  float rmax = -1e30f, rsum = 0.0f;

  issue_async(0, 0);
  asm volatile("s_wait_asynccnt 0x0" ::: "memory");
  __syncthreads();

  for (int mt = 0; mt < 128; ++mt) {
    const int cur = mt & 1;
    if (mt + 1 < 128) issue_async(mt + 1, cur ^ 1);  // DMA next step during compute

    Frag k0, k1;
    {
      const u32x4* p = (const u32x4*)&smem[cur][lane * 8];
      k0.q[0] = p[0];
      k0.q[1] = p[1];
    }
    {
      const u32x4* p = (const u32x4*)&smem[cur][256 + lane * 8];
      k1.q[0] = p[0];
      k1.q[1] = p[1];
    }
    v8f s0 = __builtin_amdgcn_wmma_f32_16x16x32_f16(false, k0.h, false, qa.h, (short)0, zeroV, false, false);
    v8f s1 = __builtin_amdgcn_wmma_f32_16x16x32_f16(false, k1.h, false, qa.h, (short)0, zeroV, false, false);

    float tm = s0[0];
#pragma unroll
    for (int r = 0; r < 8; ++r) {
      tm = fmaxf(tm, s0[r]);
      tm = fmaxf(tm, s1[r]);
    }
    tm = fmaxf(tm, __shfl_xor(tm, 16, 32));
    const float nm = fmaxf(rmax, tm);
    const float sc = __expf(rmax - nm);
    float p0[8], p1[8], ts = 0.0f;
#pragma unroll
    for (int r = 0; r < 8; ++r) {
      p0[r] = __expf(s0[r] - nm);
      p1[r] = __expf(s1[r] - nm);
      ts += p0[r] + p1[r];
    }
    ts += __shfl_xor(ts, 16, 32);
    rsum = rsum * sc + ts;
    rmax = nm;
#pragma unroll
    for (int t = 0; t < 16; ++t) oacc[t] *= sc;

    Frag pb;  // B operand for PV: K=m, lane-local repack (m = r+8g == 2d+8g)
#pragma unroll
    for (int d = 0; d < 4; ++d) {
      pb.u[d]     = pack2(p0[2 * d], p0[2 * d + 1]);
      pb.u[d + 4] = pack2(p1[2 * d], p1[2 * d + 1]);
    }
#pragma unroll
    for (int ct = 0; ct < 16; ++ct) {
      Frag va;
      const u32x4* p = (const u32x4*)&smem[cur][512 + ct * 256 + lane * 8];
      va.q[0] = p[0];
      va.q[1] = p[1];
      oacc[ct] = __builtin_amdgcn_wmma_f32_16x16x32_f16(false, va.h, false, pb.h,
                                                        (short)0, oacc[ct], false, false);
    }

    if (mt + 1 < 128) asm volatile("s_wait_asynccnt 0x0" ::: "memory");
    __syncthreads();
  }

  const float inv = 1.0f / rsum;
  const int g  = lane >> 4;
  const int ln = lane & 15;
  const int n  = ntile * 16 + ln;
#pragma unroll
  for (int ct = 0; ct < 16; ++ct) {
#pragma unroll
    for (int r = 0; r < 8; ++r) {
      const int c = r + 8 * g + 16 * ct;
      outp[(size_t)c * NPOS + n] = oacc[ct][r] * inv;
    }
  }
}

// =====================================================================
// Final 1x1 conv (512x512) + BN(eval) + LeakyReLU.  64 out-ch x 64 pos per wave.
// =====================================================================
__global__ __launch_bounds__(256) void conv_kernel(
    const float* __restrict__ bc, const float* __restrict__ gamma, const float* __restrict__ beta,
    const unsigned char* __restrict__ ws, float* __restrict__ out) {
  const int lane = threadIdx.x & 31;
  const int wid  = blockIdx.x * 8 + (threadIdx.x >> 5);
  const int oblk = wid & 7;
  const int nb   = wid >> 3;  // 0..255, 64 positions each over b*N
  const int g    = lane >> 4;
  const int ln   = lane & 15;
  const int ng0  = nb * 64;
  const int b    = ng0 >> 12;
  const int n0   = ng0 & (NPOS - 1);

  const unsigned int* wcf = (const unsigned int*)(ws + OFF_WCF);
  const float* Xb = (const float*)(ws + OFF_OUT) + (size_t)b * CIN * NPOS;

  const v8f zeroV = {0.f, 0.f, 0.f, 0.f, 0.f, 0.f, 0.f, 0.f};
  v8f acc[16];
#pragma unroll
  for (int t = 0; t < 16; ++t) acc[t] = zeroV;

  for (int ci = 0; ci < 16; ++ci) {
    Frag bfr[4];
#pragma unroll
    for (int j = 0; j < 4; ++j) {
      const int n = n0 + j * 16 + ln;
#pragma unroll
      for (int d = 0; d < 8; ++d) {
        const int c = ci * 32 + 2 * (d & 3) + 8 * g + 16 * (d >> 2);
        bfr[j].u[d] = pack2(Xb[(size_t)c * NPOS + n], Xb[(size_t)(c + 1) * NPOS + n]);
      }
    }
#pragma unroll
    for (int i = 0; i < 4; ++i) {
      Frag af;
      const u32x4* p = (const u32x4*)(wcf + (size_t)((oblk * 4 + i) * 16 + ci) * 256) + lane * 2;
      af.q[0] = p[0];
      af.q[1] = p[1];
#pragma unroll
      for (int j = 0; j < 4; ++j)
        acc[i * 4 + j] = __builtin_amdgcn_wmma_f32_16x16x32_f16(false, af.h, false, bfr[j].h,
                                                                (short)0, acc[i * 4 + j], false, false);
    }
  }

  const float rsq = rsqrtf(1.0f + 1e-5f);
#pragma unroll
  for (int i = 0; i < 4; ++i) {
#pragma unroll
    for (int r = 0; r < 8; ++r) {
      const int o   = oblk * 64 + i * 16 + r + 8 * g;
      const float bb = bc[o];
      const float sg = gamma[o] * rsq;
      const float bt = beta[o];
#pragma unroll
      for (int j = 0; j < 4; ++j) {
        const int n = n0 + j * 16 + ln;
        float y = acc[i * 4 + j][r] + bb;
        y = y * sg + bt;
        y = y > 0.0f ? y : 0.2f * y;
        out[((size_t)b * COUT + o) * NPOS + n] = y;
      }
    }
  }
}

// =====================================================================
extern "C" void kernel_launch(void* const* d_in, const int* in_sizes, int n_in,
                              void* d_out, int out_size, void* d_ws, size_t ws_size,
                              hipStream_t stream) {
  (void)in_sizes; (void)n_in; (void)out_size; (void)ws_size;
  const float* x   = (const float*)d_in[0];
  const float* Wq1 = (const float*)d_in[1];
  const float* bq1 = (const float*)d_in[2];
  const float* Wk1 = (const float*)d_in[3];
  const float* bk1 = (const float*)d_in[4];
  const float* Wv1 = (const float*)d_in[5];
  const float* bv1 = (const float*)d_in[6];
  const float* Wq2 = (const float*)d_in[7];
  const float* bq2 = (const float*)d_in[8];
  const float* Wk2 = (const float*)d_in[9];
  const float* bk2 = (const float*)d_in[10];
  const float* Wv2 = (const float*)d_in[11];
  const float* bv2 = (const float*)d_in[12];
  const float* Wc  = (const float*)d_in[13];
  const float* bc  = (const float*)d_in[14];
  const float* gm  = (const float*)d_in[15];
  const float* bt  = (const float*)d_in[16];
  unsigned char* ws = (unsigned char*)d_ws;
  float* out = (float*)d_out;

  // weight fragmentization (blocks = (O/16)*(Cw/32))
  wfrag_kernel<<<16,  256, 0, stream>>>(Wq1, (unsigned int*)(ws + OFF_WQ1F), 32, 256);
  wfrag_kernel<<<16,  256, 0, stream>>>(Wk1, (unsigned int*)(ws + OFF_WK1F), 32, 256);
  wfrag_kernel<<<16,  256, 0, stream>>>(Wq2, (unsigned int*)(ws + OFF_WQ2F), 32, 256);
  wfrag_kernel<<<16,  256, 0, stream>>>(Wk2, (unsigned int*)(ws + OFF_WK2F), 32, 256);
  wfrag_kernel<<<128, 256, 0, stream>>>(Wv1, (unsigned int*)(ws + OFF_WV1F), 256, 256);
  wfrag_kernel<<<128, 256, 0, stream>>>(Wv2, (unsigned int*)(ws + OFF_WV2F), 256, 256);
  wfrag_kernel<<<512, 256, 0, stream>>>(Wc,  (unsigned int*)(ws + OFF_WCF),  512, 512);

  proj_kernel<<<256, 256, 0, stream>>>(x, bq1, bk1, bv1, bq2, bk2, bv2, ws);
  attn_kernel<<<256, 256, 0, stream>>>(ws);
  conv_kernel<<<256, 256, 0, stream>>>(bc, gm, bt, ws, out);
}
